// SeperationBlock_Stateful_31739808317738
// MI455X (gfx1250) — matrix operand order repaired and checked by reading, not compiled, not run on hardware
//
#include <hip/hip_runtime.h>
#include <math.h>

// ---------------------------------------------------------------------------
// SeperationBlock (2-layer LSTM + dense sigmoid mask) for MI455X / gfx1250.
// Plan:
//   prep_weights : f32 -> bf16 weight conversion (+ zero padding)
//   proj1        : xp = x @ W_ih1^T + b_ih1 + b_hh1          (WMMA bf16 GEMM)
//   lstm_rec     : persistent per-batch-slice recurrence, W_hh in LDS (WMMA)
//                  W_hh staged by the Tensor Data Mover (tensor_load_to_lds)
//                  with HW padding for bank-conflict-free fragments
//   proj2        : xp = x1 @ W_ih2^T + b_ih2 + b_hh2         (WMMA bf16 GEMM)
//   lstm_rec     : layer 2
//   dense        : mask = sigmoid(x2 @ W_d^T + b_d)          (WMMA bf16 GEMM)
// Workspace (~329 MB): xp f32 [128000,512] (reused), x1/x2 bf16, bf16 weights.
// ---------------------------------------------------------------------------

typedef __bf16 bf16;
typedef __attribute__((ext_vector_type(16))) __bf16 v16bf;
typedef __attribute__((ext_vector_type(8)))  __bf16 bf8v;
typedef __attribute__((ext_vector_type(8)))  float  v8f;
typedef __attribute__((ext_vector_type(4)))  unsigned int u32x4;
typedef __attribute__((ext_vector_type(8)))  int i32x8;
typedef __attribute__((ext_vector_type(4)))  int i32x4;

#define Dd   257
#define Hh   128
#define Nn   64
#define Tt   2000
#define Gg   512          // 4*H
#define RR   (Nn * Tt)    // 128000 rows (n*T + t)
#define KP1  288          // 257 padded to 9 k32 chunks
#define DP   272          // 257 padded to 17 n16 tiles

// LDS strides (padded for bank-conflict-free fragment gathers)
#define SWS  136          // W_hh row stride in halves (272 B, 16B aligned)
#define SHS  136          // h row stride in halves
#define SGS  520          // gate row stride in floats
#define REC_SMEM (Gg*SWS*2 + 16*SHS*2 + 16*Hh*4 + 16*SGS*4)  // 185088 B

__device__ __forceinline__ v16bf mk16(bf8v lo, bf8v hi) {
  return __builtin_shufflevector(lo, hi, 0,1,2,3,4,5,6,7,8,9,10,11,12,13,14,15);
}

__device__ __forceinline__ v8f wmma_bf16(v16bf a, v16bf b, v8f c) {
  // (neg_a, A, neg_b, B, c_mod, C, reuse_a, reuse_b)
  return __builtin_amdgcn_wmma_f32_16x16x32_bf16(false, a, false, b, (short)0, c,
                                                 false, false);
}

__device__ __forceinline__ float sigm(float x) { return 1.f / (1.f + __expf(-x)); }

// ---------------------------------------------------------------------------
// Weight conversion f32 -> bf16 with zero padding.
// ---------------------------------------------------------------------------
__global__ __launch_bounds__(256) void prep_weights(
    const float* __restrict__ Wih1, const float* __restrict__ Whh1,
    const float* __restrict__ Wih2, const float* __restrict__ Whh2,
    const float* __restrict__ Wd,
    bf16* __restrict__ Wih1b, bf16* __restrict__ Whh1b,
    bf16* __restrict__ Wih2b, bf16* __restrict__ Whh2b,
    bf16* __restrict__ Wdb) {
  int idx = blockIdx.x * 256 + threadIdx.x;
  const int s1 = Gg * KP1;   // 147456
  const int s2 = Gg * Hh;    // 65536
  const int s5 = DP * Hh;    // 34816
  if (idx < s1) {
    int g = idx / KP1, k = idx % KP1;
    Wih1b[idx] = (bf16)((k < Dd) ? Wih1[g * Dd + k] : 0.f);
    return;
  }
  idx -= s1;
  if (idx < s2) { Whh1b[idx] = (bf16)Whh1[idx]; return; }
  idx -= s2;
  if (idx < s2) { Wih2b[idx] = (bf16)Wih2[idx]; return; }
  idx -= s2;
  if (idx < s2) { Whh2b[idx] = (bf16)Whh2[idx]; return; }
  idx -= s2;
  if (idx < s5) {
    int d = idx / Hh, k = idx % Hh;
    Wdb[idx] = (bf16)((d < Dd) ? Wd[d * Hh + k] : 0.f);
  }
}

// ---------------------------------------------------------------------------
// proj1: xp[r][g] = sum_k x[r][k]*W_ih1[g][k] + b_ih1[g] + b_hh1[g]
// Block = 256 thr (8 waves), covers 128 rows x 32 gate cols; wave: 2 tiles.
// ---------------------------------------------------------------------------
__global__ __launch_bounds__(256) void proj1_kernel(
    const float* __restrict__ x, const bf16* __restrict__ Wb,
    const float* __restrict__ bi, const float* __restrict__ bh,
    float* __restrict__ xp) {
  const int wave = threadIdx.x >> 5, lane = threadIdx.x & 31;
  const int lrow = lane & 15, hi = lane >> 4;
  const int r0 = blockIdx.x * 128 + wave * 16;
  const int g0 = blockIdx.y * 32;
  const float* xr = x + (size_t)(r0 + lrow) * Dd;
  const bf16* w0r = Wb + (size_t)(g0 + lrow) * KP1 + hi * 8;
  const bf16* w1r = Wb + (size_t)(g0 + 16 + lrow) * KP1 + hi * 8;
  v8f acc0 = {0.f,0.f,0.f,0.f,0.f,0.f,0.f,0.f};
  v8f acc1 = acc0;
  for (int kb = 0; kb < 9; ++kb) {
    const int ka = kb * 32 + hi * 8;
    v16bf a;
#pragma unroll
    for (int j = 0; j < 8; ++j) {
      const int k1 = ka + j, k2 = ka + 16 + j;
      a[j]     = (bf16)((k1 < Dd) ? xr[k1] : 0.f);
      a[8 + j] = (bf16)((k2 < Dd) ? xr[k2] : 0.f);
    }
    const bf16* p0 = w0r + kb * 32;
    const bf16* p1 = w1r + kb * 32;
    v16bf b0 = mk16(*(const bf8v*)p0, *(const bf8v*)(p0 + 16));
    v16bf b1 = mk16(*(const bf8v*)p1, *(const bf8v*)(p1 + 16));
    acc0 = wmma_bf16(a, b0, acc0);
    acc1 = wmma_bf16(a, b1, acc1);
  }
  const int mrow = r0 + hi * 8;
  const int gc0 = g0 + lrow, gc1 = g0 + 16 + lrow;
  const float bb0 = bi[gc0] + bh[gc0];
  const float bb1 = bi[gc1] + bh[gc1];
#pragma unroll
  for (int vr = 0; vr < 8; ++vr) {
    const size_t r = (size_t)(mrow + vr);
    xp[r * Gg + gc0] = acc0[vr] + bb0;
    xp[r * Gg + gc1] = acc1[vr] + bb1;
  }
}

// ---------------------------------------------------------------------------
// proj2: same shape, but A is bf16 [R,128] (K = 128, no padding).
// ---------------------------------------------------------------------------
__global__ __launch_bounds__(256) void proj2_kernel(
    const bf16* __restrict__ ab, const bf16* __restrict__ Wb,
    const float* __restrict__ bi, const float* __restrict__ bh,
    float* __restrict__ xp) {
  const int wave = threadIdx.x >> 5, lane = threadIdx.x & 31;
  const int lrow = lane & 15, hi = lane >> 4;
  const int r0 = blockIdx.x * 128 + wave * 16;
  const int g0 = blockIdx.y * 32;
  const bf16* ar = ab + (size_t)(r0 + lrow) * Hh + hi * 8;
  const bf16* w0r = Wb + (size_t)(g0 + lrow) * Hh + hi * 8;
  const bf16* w1r = Wb + (size_t)(g0 + 16 + lrow) * Hh + hi * 8;
  v8f acc0 = {0.f,0.f,0.f,0.f,0.f,0.f,0.f,0.f};
  v8f acc1 = acc0;
#pragma unroll
  for (int kb = 0; kb < 4; ++kb) {
    const bf16* pa = ar + kb * 32;
    v16bf a = mk16(*(const bf8v*)pa, *(const bf8v*)(pa + 16));
    const bf16* p0 = w0r + kb * 32;
    const bf16* p1 = w1r + kb * 32;
    v16bf b0 = mk16(*(const bf8v*)p0, *(const bf8v*)(p0 + 16));
    v16bf b1 = mk16(*(const bf8v*)p1, *(const bf8v*)(p1 + 16));
    acc0 = wmma_bf16(a, b0, acc0);
    acc1 = wmma_bf16(a, b1, acc1);
  }
  const int mrow = r0 + hi * 8;
  const int gc0 = g0 + lrow, gc1 = g0 + 16 + lrow;
  const float bb0 = bi[gc0] + bh[gc0];
  const float bb1 = bi[gc1] + bh[gc1];
#pragma unroll
  for (int vr = 0; vr < 8; ++vr) {
    const size_t r = (size_t)(mrow + vr);
    xp[r * Gg + gc0] = acc0[vr] + bb0;
    xp[r * Gg + gc1] = acc1[vr] + bb1;
  }
}

// ---------------------------------------------------------------------------
// Persistent LSTM recurrence. Grid = 4 WGs x 1024 threads; WG b owns batch
// rows [16b, 16b+16). W_hh (bf16) + h (bf16) + c (f32) + gates (f32) live in
// LDS for all 2000 steps. W_hh is staged by the Tensor Data Mover with
// hardware row padding (64 dwords data + 4 dwords pad == SWS stride).
// ---------------------------------------------------------------------------
__global__ __launch_bounds__(1024, 1) void lstm_rec(
    const float* __restrict__ xp, const bf16* __restrict__ Whhb,
    const float* __restrict__ h0, const float* __restrict__ c0,
    bf16* __restrict__ xoutb, float* __restrict__ hT, float* __restrict__ cT) {
  extern __shared__ char smem[];
  bf16*  sW = (bf16*)smem;                              // 512*SWS halves
  bf16*  sh = (bf16*)(smem + Gg * SWS * 2);             // 16*SHS halves
  float* sc = (float*)(smem + Gg * SWS * 2 + 16 * SHS * 2);       // 16*128
  float* sg = (float*)(smem + Gg * SWS * 2 + 16 * SHS * 2 + 16 * Hh * 4);

  const int tid = threadIdx.x;
  const int nbase = blockIdx.x * 16;

#if defined(__has_builtin) && __has_builtin(__builtin_amdgcn_tensor_load_to_lds) && \
    __has_builtin(__builtin_amdgcn_s_wait_tensorcnt)
  // --- TDM: one descriptor moves all of W_hh (512x128 bf16 = 512 rows of
  //     64 dwords) into LDS, inserting 4 dwords of pad per row (stride 272B).
  if (tid < 32) {  // wave 0 issues the DMA
    const unsigned long long ga = (unsigned long long)(uintptr_t)Whhb;
    const unsigned ldsa = (unsigned)(uintptr_t)(void*)sW;  // LDS addr[31:0]
    u32x4 g0;
    g0.x = 1u;                                     // count=1, user descriptor
    g0.y = ldsa;                                   // lds_addr
    g0.z = (unsigned)(ga & 0xFFFFFFFFu);           // global_addr[31:0]
    g0.w = (unsigned)((ga >> 32) & 0x01FFFFFFu)    // global_addr[56:32]
         | (2u << 30);                             // type = 2 ("image")
    i32x8 g1;
    g1[0] = (int)((2u << 16)      // data_size = 4 B
                | (1u << 20)      // pad_enable
                | (5u << 22)      // pad_interval: 64 dwords (one weight row)
                | (3u << 25));    // pad_amount: 4 dwords (16 B)
    g1[1] = (int)(64u << 16);     // atomic_barrier_addr=0 | tensor_dim0=64
    g1[2] = (int)(512u << 16);    // tensor_dim0.hi=0     | tensor_dim1=512
    g1[3] = (int)(64u << 16);     // tensor_dim1.hi=0     | tile_dim0=64
    g1[4] = (int)512;             // tile_dim1=512        | tile_dim2=0
    g1[5] = 64;                   // tensor_dim0_stride = 64 dwords
    g1[6] = 0;                    // stride0.hi=0 | tensor_dim1_stride.lo=0
    g1[7] = 0;
    const i32x4 z4 = {0, 0, 0, 0};
    const i32x8 z8 = {0, 0, 0, 0, 0, 0, 0, 0};
    // 6-arg form (amdgpu-toolchain / clang-23 per probe):
    //   (u32x4 g0, i32x8 g1, i32x4 g2, i32x4 g3, i32x8 pad, i32 cpol)
    __builtin_amdgcn_tensor_load_to_lds(g0, g1, z4, z4, z8, 0);
    __builtin_amdgcn_s_wait_tensorcnt((unsigned short)0);
  }
#else
  for (int i8 = tid; i8 < Gg * Hh / 8; i8 += 1024) {
    const int row = i8 >> 4, c8 = (i8 & 15) << 3;
    *(bf8v*)(sW + row * SWS + c8) = *(const bf8v*)(Whhb + row * Hh + c8);
  }
#endif

  // Stage h0 (bf16) and c0 (f32) while the TDM streams weights.
  for (int i = tid; i < 16 * Hh; i += 1024) {
    const int m = i >> 7, j = i & 127;
    sh[m * SHS + j] = (bf16)h0[(nbase + m) * Hh + j];
    sc[i] = c0[(nbase + m) * Hh + j];
  }
  __syncthreads();

  const int wave = tid >> 5, lane = tid & 31;
  const int n0 = wave * 16;           // gate column tile
  const int lrow = lane & 15, hi = lane >> 4;

  for (int t = 0; t < Tt; ++t) {
    // --- gates tile: h @ W_hh^T ---
    v8f acc = {0.f,0.f,0.f,0.f,0.f,0.f,0.f,0.f};
#pragma unroll
    for (int kb = 0; kb < 4; ++kb) {
      const bf16* ap = sh + lrow * SHS + kb * 32 + hi * 8;
      v16bf a = mk16(*(const bf8v*)ap, *(const bf8v*)(ap + 16));
      const bf16* bp = sW + (n0 + lrow) * SWS + kb * 32 + hi * 8;
      v16bf b = mk16(*(const bf8v*)bp, *(const bf8v*)(bp + 16));
      acc = wmma_bf16(a, b, acc);
    }
    // --- add xp and spill gate tile to LDS ---
    const int g = n0 + lrow;
    const int mbase = hi * 8;
#pragma unroll
    for (int vr = 0; vr < 8; ++vr) {
      const int m = mbase + vr;
      const size_t r = (size_t)(nbase + m) * Tt + t;
      sg[m * SGS + g] = acc[vr] + xp[r * Gg + g];
    }
    __syncthreads();
    // --- elementwise cell update (2048 elems / 1024 threads) ---
#pragma unroll
    for (int it = 0; it < 2; ++it) {
      const int idx = tid + it * 1024;
      const int m = idx >> 7, j = idx & 127;
      const float gi = sg[m * SGS + j];
      const float gf = sg[m * SGS + 128 + j];
      const float gc = sg[m * SGS + 256 + j];
      const float go = sg[m * SGS + 384 + j];
      const float cc = sigm(gf) * sc[idx] + sigm(gi) * tanhf(gc);
      const float hh = sigm(go) * tanhf(cc);
      sc[idx] = cc;
      sh[m * SHS + j] = (bf16)hh;
      const size_t r = (size_t)(nbase + m) * Tt + t;
      xoutb[r * Hh + j] = (bf16)hh;
      if (t == Tt - 1) {
        hT[(nbase + m) * Hh + j] = hh;
        cT[(nbase + m) * Hh + j] = cc;
      }
    }
    __syncthreads();
  }
}

// ---------------------------------------------------------------------------
// dense: mask[r][d] = sigmoid(sum_k x2[r][k]*W_d[d][k] + b_d[d]); D padded 272.
// ---------------------------------------------------------------------------
__global__ __launch_bounds__(256) void dense_kernel(
    const bf16* __restrict__ ab, const bf16* __restrict__ Wb,
    const float* __restrict__ bd, float* __restrict__ out) {
  const int wave = threadIdx.x >> 5, lane = threadIdx.x & 31;
  const int lrow = lane & 15, hi = lane >> 4;
  const int r0 = blockIdx.x * 128 + wave * 16;
  const int d0 = blockIdx.y * 16;
  const bf16* ar = ab + (size_t)(r0 + lrow) * Hh + hi * 8;
  const bf16* wr = Wb + (size_t)(d0 + lrow) * Hh + hi * 8;
  v8f acc = {0.f,0.f,0.f,0.f,0.f,0.f,0.f,0.f};
#pragma unroll
  for (int kb = 0; kb < 4; ++kb) {
    const bf16* pa = ar + kb * 32;
    v16bf a = mk16(*(const bf8v*)pa, *(const bf8v*)(pa + 16));
    const bf16* pw = wr + kb * 32;
    v16bf b = mk16(*(const bf8v*)pw, *(const bf8v*)(pw + 16));
    acc = wmma_bf16(a, b, acc);
  }
  const int d = d0 + lrow;
  if (d < Dd) {
    const float bb = bd[d];
    const int mrow = r0 + hi * 8;
#pragma unroll
    for (int vr = 0; vr < 8; ++vr) {
      out[(size_t)(mrow + vr) * Dd + d] = sigm(acc[vr] + bb);
    }
  }
}

// ---------------------------------------------------------------------------
extern "C" void kernel_launch(void* const* d_in, const int* in_sizes, int n_in,
                              void* d_out, int out_size, void* d_ws, size_t ws_size,
                              hipStream_t stream) {
  const float* x    = (const float*)d_in[0];
  const float* h1i  = (const float*)d_in[1];
  const float* c1i  = (const float*)d_in[2];
  const float* h2i  = (const float*)d_in[3];
  const float* c2i  = (const float*)d_in[4];
  const float* Wih1 = (const float*)d_in[5];
  const float* Whh1 = (const float*)d_in[6];
  const float* bih1 = (const float*)d_in[7];
  const float* bhh1 = (const float*)d_in[8];
  const float* Wih2 = (const float*)d_in[9];
  const float* Whh2 = (const float*)d_in[10];
  const float* bih2 = (const float*)d_in[11];
  const float* bhh2 = (const float*)d_in[12];
  const float* Wd   = (const float*)d_in[13];
  const float* bd   = (const float*)d_in[14];

  float* mask   = (float*)d_out;                       // [N,T,D]
  float* states = mask + (size_t)RR * Dd;              // [4,N,H]: h1,c1,h2,c2

  // workspace carve (needs ~329 MB)
  float* xp   = (float*)d_ws;                          // [R,512] (reused)
  bf16* x1b   = (bf16*)(xp + (size_t)RR * Gg);         // [R,128]
  bf16* x2b   = x1b + (size_t)RR * Hh;                 // [R,128]
  bf16* Wih1b = x2b + (size_t)RR * Hh;                 // [512,288]
  bf16* Whh1b = Wih1b + (size_t)Gg * KP1;              // [512,128]
  bf16* Wih2b = Whh1b + (size_t)Gg * Hh;               // [512,128]
  bf16* Whh2b = Wih2b + (size_t)Gg * Hh;               // [512,128]
  bf16* Wdb   = Whh2b + (size_t)Gg * Hh;               // [272,128]

  (void)in_sizes; (void)n_in; (void)out_size; (void)ws_size;

  (void)hipFuncSetAttribute((const void*)lstm_rec,
                            hipFuncAttributeMaxDynamicSharedMemorySize, REC_SMEM);

  // weight prep: 147456 + 3*65536 + 34816 = 378880 elems
  prep_weights<<<378880 / 256, 256, 0, stream>>>(Wih1, Whh1, Wih2, Whh2, Wd,
                                                 Wih1b, Whh1b, Wih2b, Whh2b, Wdb);

  dim3 pg(RR / 128, Gg / 32);  // 1000 x 16
  proj1_kernel<<<pg, 256, 0, stream>>>(x, Wih1b, bih1, bhh1, xp);
  lstm_rec<<<Nn / 16, 1024, REC_SMEM, stream>>>(xp, Whh1b, h1i, c1i, x1b,
                                                states + 0 * Nn * Hh,
                                                states + 1 * Nn * Hh);
  proj2_kernel<<<pg, 256, 0, stream>>>(x1b, Wih2b, bih2, bhh2, xp);
  lstm_rec<<<Nn / 16, 1024, REC_SMEM, stream>>>(xp, Whh2b, h2i, c2i, x2b,
                                                states + 2 * Nn * Hh,
                                                states + 3 * Nn * Hh);
  dim3 dg(RR / 128, DP / 16);  // 1000 x 17
  dense_kernel<<<dg, 256, 0, stream>>>(x2b, Wdb, bd, mask);
}
